// EncoderSum_84104049590408
// MI455X (gfx1250) — compile-verified, matrix operand order checked by reading, not run on hardware
//
#include <hip/hip_runtime.h>
#include <hip/hip_bf16.h>

// ---------------------------------------------------------------------------
// GraphCast grid->mesh encoder, MI455X (gfx1250, wave32, WMMA bf16).
//
// Pipeline (all on `stream`):
//   1) transpose+convert 6 weight matrices fp32[K,N] -> bf16 Wt[N,K] in d_ws
//   2) memset agg region (mesh half of d_out) to zero
//   3) edge kernel   : MLP(concat(efeat, grid[src], mesh[dst])) -> atomicAdd agg[dst]
//   4) grid kernel   : grid_out = grid + MLP(grid)
//   5) mesh kernel   : mesh_out = mesh + MLP(concat(agg, mesh))   (in-place over agg)
// ---------------------------------------------------------------------------

typedef __bf16 bf16_t;
typedef __attribute__((ext_vector_type(16))) __bf16 v16bf;
typedef __attribute__((ext_vector_type(8)))  __bf16 v8bf;
typedef __attribute__((ext_vector_type(8)))  float  v8f;

#define GRID_N 65536
#define MESH_N 40962
#define E_N    131072
#define DD     512
#define HH     512
#define HSTRIDE 520   // LDS row stride in bf16 elems (1040B -> rows spread over banks)

__device__ inline v8f wmma_bf16(v16bf a, v16bf b, v8f c) {
    // D = A(16x32) * B(32x16) + C, f32 accumulate
    return __builtin_amdgcn_wmma_f32_16x16x32_bf16(
        /*neg_a=*/false, a, /*neg_b=*/false, b,
        /*c_mod=*/(short)0, c, /*reuse_a=*/false, /*reuse_b=*/false);
}

// Per-lane 16-bit A-matrix fragment pattern (ISA 7.12.2): lane holds 8
// consecutive K at kBase and 8 consecutive K at kBase+16. Caller passes a
// pointer at (row, kBase).
__device__ inline v16bf frag_from_f32(const float* __restrict__ p) {
    union { v16bf v; __bf16 e[16]; } u;
    float4 a = *reinterpret_cast<const float4*>(p);
    float4 b = *reinterpret_cast<const float4*>(p + 4);
    float4 c = *reinterpret_cast<const float4*>(p + 16);
    float4 d = *reinterpret_cast<const float4*>(p + 20);
    u.e[0] = (__bf16)a.x;  u.e[1] = (__bf16)a.y;  u.e[2] = (__bf16)a.z;  u.e[3] = (__bf16)a.w;
    u.e[4] = (__bf16)b.x;  u.e[5] = (__bf16)b.y;  u.e[6] = (__bf16)b.z;  u.e[7] = (__bf16)b.w;
    u.e[8] = (__bf16)c.x;  u.e[9] = (__bf16)c.y;  u.e[10] = (__bf16)c.z; u.e[11] = (__bf16)c.w;
    u.e[12] = (__bf16)d.x; u.e[13] = (__bf16)d.y; u.e[14] = (__bf16)d.z; u.e[15] = (__bf16)d.w;
    return u.v;
}

__device__ inline v16bf frag_from_bf16(const bf16_t* __restrict__ p) {
    union { v16bf v; v8bf h[2]; } u;
    u.h[0] = *reinterpret_cast<const v8bf*>(p);
    u.h[1] = *reinterpret_cast<const v8bf*>(p + 16);
    return u.v;
}

// ---------------------------------------------------------------------------
// Fused 2-layer MLP + LayerNorm tile kernel.
//   MODE 0: edge   K1=1536  A = concat(A0[row], A1[src[row]], A2[dst[row]])
//                  epilogue: atomicAdd into agg[dst[row]]
//   MODE 1: grid   K1=512   A = A0[row]; epilogue: out = resid + y
//   MODE 2: mesh   K1=1024  A = concat(A0[row], A1[row]); epilogue: out = resid + y
// Block = 256 threads (8 waves), M-tile = 32 rows, wave w owns N cols [64w,64w+64).
// GEMM1 runs as K1/512 compile-time passes with fixed base pointers so the
// inner loop is branch-free (pure load/cvt/wmma).
// ---------------------------------------------------------------------------
template<int MODE, int K1>
__global__ __launch_bounds__(256) void fused_mlp(
    const float* __restrict__ A0, const float* __restrict__ A1,
    const float* __restrict__ A2,
    const int* __restrict__ srcIdx, const int* __restrict__ dstIdx,
    const bf16_t* __restrict__ W1t, const float* __restrict__ b1,
    const bf16_t* __restrict__ W2t, const float* __restrict__ b2,
    const float* __restrict__ gw, const float* __restrict__ bw,
    const float* __restrict__ resid, float* __restrict__ out,
    float* __restrict__ agg, int nRows)
{
    constexpr int NSEG = K1 / 512;

    __shared__ __align__(16) bf16_t hbuf[32 * HSTRIDE];
    __shared__ float lnSum[32];
    __shared__ float lnSq[32];
    __shared__ int   dstSh[32];

    const int tid = threadIdx.x;
    const int w   = tid >> 5;       // wave 0..7
    const int l   = tid & 31;       // lane
    const int lm  = l & 15;
    const int hi  = l >> 4;         // lane-half select in fragment layout
    const int blockRow = blockIdx.x * 32;

    if (tid < 32) {
        lnSum[tid] = 0.f; lnSq[tid] = 0.f;
        if (MODE == 0) dstSh[tid] = dstIdx[blockRow + tid];
    }
    __syncthreads();

    // Per-lane, per-segment A base pointers (A-fragment row = lm in each subtile)
    const float* aBase[2][NSEG];
#pragma unroll
    for (int mi = 0; mi < 2; ++mi) {
        int r = blockRow + mi * 16 + lm;
        if (MODE == 2) r = (r < nRows) ? r : (nRows - 1);
        if (MODE == 0) {
            aBase[mi][0] = A0 + (size_t)r * DD + hi * 8;
            aBase[mi][1] = A1 + (size_t)srcIdx[r] * DD + hi * 8;
            aBase[mi][2] = A2 + (size_t)dstSh[mi * 16 + lm] * DD + hi * 8;
        } else if (MODE == 1) {
            aBase[mi][0] = A0 + (size_t)r * DD + hi * 8;
        } else {
            aBase[mi][0] = A0 + (size_t)r * DD + hi * 8;
            aBase[mi][1] = A1 + (size_t)r * DD + hi * 8;
        }
    }

    // Per-lane B row pointers (output column = row of transposed weights)
    const bf16_t* b1Row[4];
    const bf16_t* b2Row[4];
#pragma unroll
    for (int ni = 0; ni < 4; ++ni) {
        int c = w * 64 + ni * 16 + lm;
        b1Row[ni] = W1t + (size_t)c * K1 + hi * 8;
        b2Row[ni] = W2t + (size_t)c * HH + hi * 8;
    }

    v8f acc[2][4];
    const v8f vzero = {0.f, 0.f, 0.f, 0.f, 0.f, 0.f, 0.f, 0.f};
#pragma unroll
    for (int mi = 0; mi < 2; ++mi)
#pragma unroll
        for (int ni = 0; ni < 4; ++ni) acc[mi][ni] = vzero;

    // ---------------- GEMM1: [32,K1] x [K1,512] (bf16 -> f32) ----------------
#pragma unroll
    for (int seg = 0; seg < NSEG; ++seg) {
        const float* a0 = aBase[0][seg];
        const float* a1 = aBase[1][seg];
        for (int k0 = 0; k0 < 512; k0 += 32) {
            v16bf af0 = frag_from_f32(a0 + k0);
            v16bf af1 = frag_from_f32(a1 + k0);
#pragma unroll
            for (int ni = 0; ni < 4; ++ni) {
                v16bf bfr = frag_from_bf16(b1Row[ni] + seg * 512 + k0);
                acc[0][ni] = wmma_bf16(af0, bfr, acc[0][ni]);
                acc[1][ni] = wmma_bf16(af1, bfr, acc[1][ni]);
            }
        }
    }

    // ------------- bias + SiLU -> bf16 hidden activations in LDS -------------
#pragma unroll
    for (int ni = 0; ni < 4; ++ni) {
        int c = w * 64 + ni * 16 + lm;
        float bias = b1[c];
#pragma unroll
        for (int mi = 0; mi < 2; ++mi)
#pragma unroll
            for (int v = 0; v < 8; ++v) {
                float x = acc[mi][ni][v] + bias;
                float s = x / (1.f + __expf(-x));       // SiLU
                int r = mi * 16 + v + 8 * hi;           // C-matrix row layout
                hbuf[r * HSTRIDE + c] = (bf16_t)s;
            }
    }
    __syncthreads();

#pragma unroll
    for (int mi = 0; mi < 2; ++mi)
#pragma unroll
        for (int ni = 0; ni < 4; ++ni) acc[mi][ni] = vzero;

    // ---------------- GEMM2: [32,512] x [512,512] (LDS A) --------------------
    {
        const bf16_t* h0 = hbuf + (0 * 16 + lm) * HSTRIDE + hi * 8;
        const bf16_t* h1 = hbuf + (1 * 16 + lm) * HSTRIDE + hi * 8;
        for (int k0 = 0; k0 < HH; k0 += 32) {
            v16bf af0 = frag_from_bf16(h0 + k0);
            v16bf af1 = frag_from_bf16(h1 + k0);
#pragma unroll
            for (int ni = 0; ni < 4; ++ni) {
                v16bf bfr = frag_from_bf16(b2Row[ni] + k0);
                acc[0][ni] = wmma_bf16(af0, bfr, acc[0][ni]);
                acc[1][ni] = wmma_bf16(af1, bfr, acc[1][ni]);
            }
        }
    }

    // ------------------ bias + LayerNorm row statistics ----------------------
#pragma unroll
    for (int ni = 0; ni < 4; ++ni) {
        int c = w * 64 + ni * 16 + lm;
        float bias = b2[c];
#pragma unroll
        for (int mi = 0; mi < 2; ++mi)
#pragma unroll
            for (int v = 0; v < 8; ++v)
                acc[mi][ni][v] = acc[mi][ni][v] + bias;
    }
#pragma unroll
    for (int mi = 0; mi < 2; ++mi)
#pragma unroll
        for (int v = 0; v < 8; ++v) {
            float sx = 0.f, sxx = 0.f;
#pragma unroll
            for (int ni = 0; ni < 4; ++ni) {
                float x = acc[mi][ni][v];
                sx += x; sxx += x * x;
            }
            int r = mi * 16 + v + 8 * hi;
            atomicAdd(&lnSum[r], sx);       // ds_add_f32
            atomicAdd(&lnSq[r], sxx);
        }
    __syncthreads();

    // ---------------- normalize + affine + epilogue --------------------------
#pragma unroll
    for (int ni = 0; ni < 4; ++ni) {
        int c = w * 64 + ni * 16 + lm;
        float gg = gw[c];
        float bb = bw[c];
#pragma unroll
        for (int mi = 0; mi < 2; ++mi)
#pragma unroll
            for (int v = 0; v < 8; ++v) {
                int r = mi * 16 + v + 8 * hi;
                float mu  = lnSum[r] * (1.f / 512.f);
                float var = lnSq[r] * (1.f / 512.f) - mu * mu;
                float y = (acc[mi][ni][v] - mu) * rsqrtf(var + 1e-5f) * gg + bb;
                int gr = blockRow + r;
                if (MODE == 0) {
                    atomicAdd(&agg[(size_t)dstSh[r] * DD + c], y);   // segment_sum
                } else if (MODE == 1) {
                    out[(size_t)gr * DD + c] = resid[(size_t)gr * DD + c] + y;
                } else {
                    if (gr < nRows)
                        out[(size_t)gr * DD + c] = resid[(size_t)gr * DD + c] + y;
                }
            }
    }
}

// fp32 W[K,N] -> bf16 Wt[N,K] (so both WMMA operands use the A-fragment layout)
__global__ void transpose_bf16(const float* __restrict__ W, bf16_t* __restrict__ Wt,
                               int K, int N)
{
    int i = blockIdx.x * 256 + threadIdx.x;
    if (i >= K * N) return;
    int k = i / N, n = i % N;
    Wt[(size_t)n * K + k] = (bf16_t)W[i];
}

extern "C" void kernel_launch(void* const* d_in, const int* in_sizes, int n_in,
                              void* d_out, int out_size, void* d_ws, size_t ws_size,
                              hipStream_t stream)
{
    const float* g2m   = (const float*)d_in[0];
    const float* gridf = (const float*)d_in[1];
    const float* meshf = (const float*)d_in[2];
    const int*   src   = (const int*)d_in[3];
    const int*   dst   = (const int*)d_in[4];
    const float* eW1 = (const float*)d_in[5];  const float* eB1 = (const float*)d_in[6];
    const float* eW2 = (const float*)d_in[7];  const float* eB2 = (const float*)d_in[8];
    const float* eG  = (const float*)d_in[9];  const float* eBt = (const float*)d_in[10];
    const float* sW1 = (const float*)d_in[11]; const float* sB1 = (const float*)d_in[12];
    const float* sW2 = (const float*)d_in[13]; const float* sB2 = (const float*)d_in[14];
    const float* sG  = (const float*)d_in[15]; const float* sBt = (const float*)d_in[16];
    const float* dW1 = (const float*)d_in[17]; const float* dB1 = (const float*)d_in[18];
    const float* dW2 = (const float*)d_in[19]; const float* dB2 = (const float*)d_in[20];
    const float* dG  = (const float*)d_in[21]; const float* dBt = (const float*)d_in[22];

    float* out  = (float*)d_out;
    float* agg  = out + (size_t)GRID_N * DD;   // mesh half of d_out doubles as agg

    // bf16 transposed weights in workspace (4.72 MB total)
    bf16_t* ws = (bf16_t*)d_ws;
    bf16_t* eW1t = ws;                      // [512][1536]
    bf16_t* eW2t = eW1t + (size_t)512 * 1536;
    bf16_t* sW1t = eW2t + (size_t)512 * 512;
    bf16_t* sW2t = sW1t + (size_t)512 * 512;
    bf16_t* dW1t = sW2t + (size_t)512 * 512; // [512][1024]
    bf16_t* dW2t = dW1t + (size_t)512 * 1024;

    auto tlaunch = [&](const float* W, bf16_t* Wt, int K, int N) {
        int n = K * N;
        transpose_bf16<<<(n + 255) / 256, 256, 0, stream>>>(W, Wt, K, N);
    };
    tlaunch(eW1, eW1t, 3 * DD, HH);
    tlaunch(eW2, eW2t, HH, DD);
    tlaunch(sW1, sW1t, DD, HH);
    tlaunch(sW2, sW2t, HH, DD);
    tlaunch(dW1, dW1t, 2 * DD, HH);
    tlaunch(dW2, dW2t, HH, DD);

    hipMemsetAsync(agg, 0, (size_t)MESH_N * DD * sizeof(float), stream);

    // Edge MLP + fused segment_sum (atomic scatter into agg)
    fused_mlp<0, 3 * DD><<<E_N / 32, 256, 0, stream>>>(
        g2m, gridf, meshf, src, dst,
        eW1t, eB1, eW2t, eB2, eG, eBt,
        nullptr, nullptr, agg, E_N);

    // Grid node MLP + residual
    fused_mlp<1, DD><<<GRID_N / 32, 256, 0, stream>>>(
        gridf, nullptr, nullptr, nullptr, nullptr,
        sW1t, sB1, sW2t, sB2, sG, sBt,
        gridf, out, nullptr, GRID_N);

    // Mesh node MLP on concat(agg, mesh) + residual; overwrites agg in place
    fused_mlp<2, 2 * DD><<<(MESH_N + 31) / 32, 256, 0, stream>>>(
        agg, meshf, nullptr, nullptr, nullptr,
        dW1t, dB1, dW2t, dB2, dG, dBt,
        meshf, agg, nullptr, MESH_N);
}
